// Attention_70832600646214
// MI455X (gfx1250) — matrix operand order repaired and checked by reading, not compiled, not run on hardware
//
#include <hip/hip_runtime.h>
#include <hip/hip_bf16.h>

typedef __bf16 bf16_t;
typedef __attribute__((ext_vector_type(16))) __bf16 v16bf;
typedef __attribute__((ext_vector_type(8)))  __bf16 v8bf;
typedef __attribute__((ext_vector_type(8)))  float  v8f;

#define B_       2
#define T_       2048
#define D_       2048
#define NH_      16
#define KH_      8
#define H_       128
#define WINDOW_  1024
#define SOFT_CAP_ 50.0f
#define INV_CAP_  0.02f
#define K_MASK_  (-2.3819763e+38f)
#define SCALE_   0.08838834764831845f   // 128^-0.5
#define LN_ROPE_OVER_HALF 0.14391156f   // ln(10000)/64

static __device__ __forceinline__ v8f wmma_bf16(v16bf a, v16bf b, v8f c) {
    return __builtin_amdgcn_wmma_f32_16x16x32_bf16(false, a, false, b, (short)0, c, false, false);
}
static __device__ __forceinline__ v16bf cat8(v8bf lo, v8bf hi) {
    return __builtin_shufflevector(lo, hi, 0,1,2,3,4,5,6,7,8,9,10,11,12,13,14,15);
}
static __device__ __forceinline__ v8bf lo8(v16bf t) {
    return __builtin_shufflevector(t, t, 0,1,2,3,4,5,6,7);
}
static __device__ __forceinline__ v8bf hi8(v16bf t) {
    return __builtin_shufflevector(t, t, 8,9,10,11,12,13,14,15);
}

// ---------------------------------------------------------------------------
// f32 -> bf16 cast (row-major copy)
// ---------------------------------------------------------------------------
__global__ void cast_f32_bf16(const float* __restrict__ in, bf16_t* __restrict__ out, int n) {
    int i = blockIdx.x * blockDim.x + threadIdx.x;
    int stride = gridDim.x * blockDim.x;
    for (; i < n; i += stride) out[i] = (bf16_t)in[i];
}

// ---------------------------------------------------------------------------
// Pack q_w [N,D,H] + kv_w [2,K,D,H] into K-major bf16 W^T [32*H][D]
// 32x32 tiled LDS transpose; block (32,8)
// ---------------------------------------------------------------------------
__global__ __launch_bounds__(256)
void pack_qkv_wt(const float* __restrict__ qw, const float* __restrict__ kvw,
                 bf16_t* __restrict__ wt) {
    __shared__ float tile[32][33];
    const int td = blockIdx.x * 32;       // D tile
    const int tc = blockIdx.y * 32;       // output-column tile (head*128+h)
    const int head  = tc >> 7;
    const int hbase = tc & 127;
    const float* src;
    if (head < 16)      src = qw  + (size_t)head * D_ * H_;
    else if (head < 24) src = kvw + (size_t)(head - 16) * D_ * H_;
    else                src = kvw + (size_t)(8 + head - 24) * D_ * H_;
    const int tx = threadIdx.x, ty = threadIdx.y;
#pragma unroll
    for (int i = 0; i < 4; ++i) {
        int d = ty + i * 8;
        tile[d][tx] = src[(size_t)(td + d) * H_ + hbase + tx];
    }
    __syncthreads();
#pragma unroll
    for (int i = 0; i < 4; ++i) {
        int c = ty + i * 8;
        wt[(size_t)(tc + c) * D_ + td + tx] = (bf16_t)tile[tx][c];
    }
}

// ---------------------------------------------------------------------------
// Generic f32 -> bf16 tiled transpose: out[c][r] = in[r][c]
// ---------------------------------------------------------------------------
__global__ __launch_bounds__(256)
void transpose_cast(const float* __restrict__ in, bf16_t* __restrict__ out, int R, int C) {
    __shared__ float tile[32][33];
    const int r0 = blockIdx.x * 32, c0 = blockIdx.y * 32;
    const int tx = threadIdx.x, ty = threadIdx.y;
#pragma unroll
    for (int i = 0; i < 4; ++i)
        tile[ty + i * 8][tx] = in[(size_t)(r0 + ty + i * 8) * C + c0 + tx];
    __syncthreads();
#pragma unroll
    for (int i = 0; i < 4; ++i)
        out[(size_t)(c0 + ty + i * 8) * R + r0 + tx] = (bf16_t)tile[tx][ty + i * 8];
}

// ---------------------------------------------------------------------------
// V transpose: qkv f32 [tok][32][128] (heads 24..31) -> vt bf16 [b*8+kh][h][t]
// ---------------------------------------------------------------------------
__global__ __launch_bounds__(256)
void transpose_v(const float* __restrict__ qkvf, bf16_t* __restrict__ vt) {
    __shared__ float tile[32][33];
    const int t0 = blockIdx.x * 32;
    const int h0 = blockIdx.y * 32;
    const int bk = blockIdx.z;            // b*8 + kh
    const int b  = bk >> 3, kh = bk & 7;
    const int tx = threadIdx.x, ty = threadIdx.y;
#pragma unroll
    for (int i = 0; i < 4; ++i) {
        int t = t0 + ty + i * 8;
        tile[ty + i * 8][tx] = qkvf[(((size_t)(b * T_ + t)) * 32 + 24 + kh) * H_ + h0 + tx];
    }
    __syncthreads();
#pragma unroll
    for (int i = 0; i < 4; ++i) {
        int h = h0 + ty + i * 8;
        vt[((size_t)bk * H_ + h) * T_ + t0 + tx] = (bf16_t)tile[tx][ty + i * 8];
    }
}

// ---------------------------------------------------------------------------
// Tiled bf16 GEMM, TN form: C[M,N] (f32) = A[M,K] * Bt[N,K]^T
// Block 256 = 8 waves (4x2); block tile 256x128; wave tile 64x64 (4x4 WMMA
// tiles, 128 acc VGPRs) -> 16 ds_load_b128 : 16 wmma per K-step.
// LDS rows padded to 40 elements (conflict-free, all chunks 16B aligned).
// ---------------------------------------------------------------------------
#define GTM 256
#define GTN 128
#define GTK 32
#define LDP 40

__global__ __launch_bounds__(256)
void gemm_bf16_tn(const bf16_t* __restrict__ A, const bf16_t* __restrict__ Bt,
                  float* __restrict__ C, int M, int N, int K) {
    __shared__ __align__(16) bf16_t As[GTM][LDP];   // 20 KB
    __shared__ __align__(16) bf16_t Bs[GTN][LDP];   // 10 KB

    const int tid  = threadIdx.x;
    const int lane = tid & 31;
    const int wid  = tid >> 5;
    const int wm   = wid & 3;          // 4 waves along M (4*64 = 256)
    const int wn   = wid >> 2;         // 2 waves along N (2*64 = 128)
    const int half = lane >> 4;
    const int l15  = lane & 15;

    const int block_m = blockIdx.y * GTM;
    const int block_n = blockIdx.x * GTN;

    v8f acc[4][4];
#pragma unroll
    for (int i = 0; i < 4; ++i)
#pragma unroll
        for (int j = 0; j < 4; ++j) acc[i][j] = (v8f){0.f,0.f,0.f,0.f,0.f,0.f,0.f,0.f};

    const int ra = tid;                 // A: one 64B row-chunk per thread
    const int rb = tid >> 1;            // B: two threads per row
    const int cb = (tid & 1) * 16;

    for (int k0 = 0; k0 < K; k0 += GTK) {
        {
            const bf16_t* ap = A + (size_t)(block_m + ra) * K + k0;
            v16bf t0 = *(const v16bf*)(ap);
            v16bf t1 = *(const v16bf*)(ap + 16);
            *(v8bf*)&As[ra][0]  = lo8(t0);
            *(v8bf*)&As[ra][8]  = hi8(t0);
            *(v8bf*)&As[ra][16] = lo8(t1);
            *(v8bf*)&As[ra][24] = hi8(t1);
        }
        {
            v16bf t = *(const v16bf*)(Bt + (size_t)(block_n + rb) * K + k0 + cb);
            *(v8bf*)&Bs[rb][cb]     = lo8(t);
            *(v8bf*)&Bs[rb][cb + 8] = hi8(t);
        }
        __syncthreads();

        v16bf afr[4];
#pragma unroll
        for (int mt = 0; mt < 4; ++mt) {
            int arow = wm * 64 + mt * 16 + l15;
            v8bf a0 = *(const v8bf*)&As[arow][half * 16];
            v8bf a1 = *(const v8bf*)&As[arow][half * 16 + 8];
            afr[mt] = cat8(a0, a1);
        }
#pragma unroll
        for (int nt = 0; nt < 4; ++nt) {
            int brow = wn * 64 + nt * 16 + l15;
            v8bf b0 = *(const v8bf*)&Bs[brow][half * 16];
            v8bf b1 = *(const v8bf*)&Bs[brow][half * 16 + 8];
            v16bf bfr = cat8(b0, b1);
#pragma unroll
            for (int mt = 0; mt < 4; ++mt)
                acc[mt][nt] = wmma_bf16(afr[mt], bfr, acc[mt][nt]);
        }
        __syncthreads();
    }

#pragma unroll
    for (int mt = 0; mt < 4; ++mt)
#pragma unroll
        for (int nt = 0; nt < 4; ++nt)
#pragma unroll
            for (int v = 0; v < 8; ++v) {
                int row = block_m + wm * 64 + mt * 16 + v + 8 * half;
                int col = block_n + wn * 64 + nt * 16 + l15;
                C[(size_t)row * N + col] = acc[mt][nt][v];
            }
}

// ---------------------------------------------------------------------------
// Fused RMS-norm + RoPE (+ scale for q).  qkv f32 [tok][32][128] ->
//   q bf16 [tok][16][128], k bf16 [tok][8][128]   (heads 0..23 only)
// grid (B*T, 24), block = 32 (one wave); lane t owns h = t,t+32,t+64,t+96
// ---------------------------------------------------------------------------
__global__ __launch_bounds__(32)
void norm_rope(const float* __restrict__ qkv,
               const float* __restrict__ qns, const float* __restrict__ kns,
               bf16_t* __restrict__ Qo, bf16_t* __restrict__ Ko) {
    const int tok  = blockIdx.x;
    const int head = blockIdx.y;
    const int t    = threadIdx.x;
    const int pos  = tok & (T_ - 1);

    const float* src = qkv + ((size_t)tok * 32 + head) * H_;
    float x0 = src[t], x1 = src[t + 32], x2 = src[t + 64], x3 = src[t + 96];

    float ss = x0 * x0 + x1 * x1 + x2 * x2 + x3 * x3;
#pragma unroll
    for (int d = 1; d < 32; d <<= 1) ss += __shfl_xor(ss, d, 32);
    float rr = rsqrtf(ss * (1.0f / 128.0f) + 1e-6f);
    const float* ns = (head < 16) ? qns : kns;
    x0 *= rr * (1.0f + ns[t]);
    x1 *= rr * (1.0f + ns[t + 32]);
    x2 *= rr * (1.0f + ns[t + 64]);
    x3 *= rr * (1.0f + ns[t + 96]);
    // rope pairs (t, t+64) and (t+32, t+96); timescale = 10000^(j/64)
    float fp = (float)pos;
    float a0 = fp * __expf(-LN_ROPE_OVER_HALF * (float)t);
    float a1 = fp * __expf(-LN_ROPE_OVER_HALF * (float)(t + 32));
    float s0 = __sinf(a0), c0 = __cosf(a0);
    float s1 = __sinf(a1), c1 = __cosf(a1);
    float o0 = x0 * c0 - x2 * s0;
    float o2 = x2 * c0 + x0 * s0;
    float o1 = x1 * c1 - x3 * s1;
    float o3 = x3 * c1 + x1 * s1;
    if (head < 16) {
        bf16_t* dst = Qo + ((size_t)tok * NH_ + head) * H_;
        dst[t]      = (bf16_t)(o0 * SCALE_);
        dst[t + 32] = (bf16_t)(o1 * SCALE_);
        dst[t + 64] = (bf16_t)(o2 * SCALE_);
        dst[t + 96] = (bf16_t)(o3 * SCALE_);
    } else {
        bf16_t* dst = Ko + ((size_t)tok * KH_ + (head - 16)) * H_;
        dst[t]      = (bf16_t)o0;
        dst[t + 32] = (bf16_t)o1;
        dst[t + 64] = (bf16_t)o2;
        dst[t + 96] = (bf16_t)o3;
    }
}

// ---------------------------------------------------------------------------
// Flash attention (sliding window + causal + tanh soft-cap), GQA G=2.
// One wave per 16-query tile; key blocks of 32; V is pre-transposed [h][t].
// ---------------------------------------------------------------------------
__global__ __launch_bounds__(256)
void flash_attn(const bf16_t* __restrict__ Q, const bf16_t* __restrict__ Kb,
                const bf16_t* __restrict__ Vt, bf16_t* __restrict__ O) {
    __shared__ __align__(16) bf16_t Plds[8][16 * 32];  // per-wave P staging (1 KB)

    const int tid  = threadIdx.x;
    const int lane = tid & 31;
    const int wid  = tid >> 5;
    const int w    = blockIdx.x * 8 + wid;

    const int qt = w & 127;           // T/16 = 128 query tiles
    const int n  = (w >> 7) & 15;     // q head
    const int b  = w >> 11;           // batch
    const int q0 = qt * 16;
    const int kh = n >> 1;            // kv head (G = 2)
    const int half = lane >> 4;
    const int l15  = lane & 15;

    // Q A-fragments over H (4 chunks of 32), contiguous 16B loads
    v16bf aq[4];
    {
        const bf16_t* qrow = Q + (((size_t)(b * T_ + q0 + l15)) * NH_ + n) * H_;
#pragma unroll
        for (int c = 0; c < 4; ++c) {
            v8bf a0 = *(const v8bf*)(qrow + c * 32 + half * 16);
            v8bf a1 = *(const v8bf*)(qrow + c * 32 + half * 16 + 8);
            aq[c] = cat8(a0, a1);
        }
    }

    v8f o_acc[8];
#pragma unroll
    for (int h = 0; h < 8; ++h) o_acc[h] = (v8f){0.f,0.f,0.f,0.f,0.f,0.f,0.f,0.f};
    float m_r[8], l_r[8];
#pragma unroll
    for (int v = 0; v < 8; ++v) { m_r[v] = -3.0e38f; l_r[v] = 0.f; }

    int kstart = q0 - (WINDOW_ - 1);
    if (kstart < 0) kstart = 0;
    kstart &= ~31;

    bf16_t* pl = &Plds[wid][0];
    const bf16_t* vbase = Vt + (size_t)(b * KH_ + kh) * H_ * T_;

    for (int kb = kstart; kb <= q0 + 15; kb += 32) {
        // ---- logits: two 16-key tiles, K-loop over H (contiguous 32B loads) ----
        v8f s0 = (v8f){0.f,0.f,0.f,0.f,0.f,0.f,0.f,0.f};
        v8f s1 = s0;
        {
            int key0 = kb + l15;        if (key0 >= T_) key0 = T_ - 1;
            int key1 = kb + 16 + l15;   if (key1 >= T_) key1 = T_ - 1;
            const bf16_t* kp0 = Kb + (((size_t)(b * T_ + key0)) * KH_ + kh) * H_ + half * 16;
            const bf16_t* kp1 = Kb + (((size_t)(b * T_ + key1)) * KH_ + kh) * H_ + half * 16;
#pragma unroll
            for (int c = 0; c < 4; ++c) {
                v16bf b0 = *(const v16bf*)(kp0 + c * 32);
                v16bf b1 = *(const v16bf*)(kp1 + c * 32);
                s0 = wmma_bf16(aq[c], b0, s0);
                s1 = wmma_bf16(aq[c], b1, s1);
            }
        }

        // ---- soft-cap + mask + online softmax (rows live in half-waves) ----
        float p0[8], p1[8];
#pragma unroll
        for (int v = 0; v < 8; ++v) {
            int qpos = q0 + v + 8 * half;
            int k0p = kb + l15, k1p = kb + 16 + l15;
            float x0 = tanhf(s0[v] * INV_CAP_) * SOFT_CAP_;
            float x1 = tanhf(s1[v] * INV_CAP_) * SOFT_CAP_;
            bool v0 = (k0p <= qpos) && (k0p > qpos - WINDOW_);
            bool v1 = (k1p <= qpos) && (k1p > qpos - WINDOW_);
            x0 = v0 ? x0 : K_MASK_;
            x1 = v1 ? x1 : K_MASK_;
            float mx = fmaxf(x0, x1);
#pragma unroll
            for (int d = 1; d < 16; d <<= 1) mx = fmaxf(mx, __shfl_xor(mx, d, 32));
            float mnew  = fmaxf(m_r[v], mx);
            float scale = __expf(m_r[v] - mnew);
            float e0 = __expf(x0 - mnew);
            float e1 = __expf(x1 - mnew);
            float rs = e0 + e1;
#pragma unroll
            for (int d = 1; d < 16; d <<= 1) rs += __shfl_xor(rs, d, 32);
            l_r[v] = l_r[v] * scale + rs;
            m_r[v] = mnew;
#pragma unroll
            for (int ht = 0; ht < 8; ++ht) o_acc[ht][v] *= scale;
            p0[v] = e0;
            p1[v] = e1;
        }

        // ---- C-layout P -> LDS -> A-fragment (wave-local) ----
#pragma unroll
        for (int v = 0; v < 8; ++v) {
            int row = v + 8 * half;
            pl[row * 32 + l15]      = (bf16_t)p0[v];
            pl[row * 32 + 16 + l15] = (bf16_t)p1[v];
        }
        asm volatile("s_wait_dscnt 0" ::: "memory");
        v8bf ap0 = *(const v8bf*)(pl + l15 * 32 + half * 16);
        v8bf ap1 = *(const v8bf*)(pl + l15 * 32 + half * 16 + 8);
        v16bf ap = cat8(ap0, ap1);

        // ---- P @ V over 8 H-tiles: contiguous 32B loads from V^T ----
#pragma unroll
        for (int ht = 0; ht < 8; ++ht) {
            const bf16_t* vp = vbase + (size_t)(ht * 16 + l15) * T_ + kb + half * 16;
            v16bf bv = *(const v16bf*)vp;   // tail over-read lands in adjacent ws region; p==0 there
            o_acc[ht] = wmma_bf16(ap, bv, o_acc[ht]);
        }
    }

    // ---- normalize + store enc (bf16) ----
#pragma unroll
    for (int v = 0; v < 8; ++v) {
        float inv = (l_r[v] > 0.f) ? (1.0f / l_r[v]) : 0.f;
#pragma unroll
        for (int ht = 0; ht < 8; ++ht) o_acc[ht][v] *= inv;
    }
#pragma unroll
    for (int ht = 0; ht < 8; ++ht)
#pragma unroll
        for (int v = 0; v < 8; ++v) {
            int row = v + 8 * half;
            int h   = ht * 16 + l15;
            O[(((size_t)(b * T_ + q0 + row)) * NH_ + n) * H_ + h] = (bf16_t)o_acc[ht][v];
        }
}

// ---------------------------------------------------------------------------
// Host launcher
// ---------------------------------------------------------------------------
extern "C" void kernel_launch(void* const* d_in, const int* in_sizes, int n_in,
                              void* d_out, int out_size, void* d_ws, size_t ws_size,
                              hipStream_t stream) {
    const float* x    = (const float*)d_in[0];
    const float* q_w  = (const float*)d_in[1];
    const float* kv_w = (const float*)d_in[2];
    const float* o_w  = (const float*)d_in[3];
    const float* qns  = (const float*)d_in[4];
    const float* kns  = (const float*)d_in[5];
    // d_in[6] segment_pos, d_in[7] attn_mask: recomputed analytically on device
    float* out = (float*)d_out;

    char* ws = (char*)d_ws;
    const size_t MB = 1u << 20;
    bf16_t* xb    = (bf16_t*)(ws + 0);          // 16 MiB: x bf16 [4096][2048]
    bf16_t* wqkvt = (bf16_t*)(ws + 16 * MB);    // 16 MiB: W^T [4096][2048]
    bf16_t* qb    = (bf16_t*)(ws + 32 * MB);    // 16 MiB: q bf16 [4096][16][128]
    bf16_t* kbuf  = (bf16_t*)(ws + 48 * MB);    //  8 MiB: k bf16 [4096][8][128]
    bf16_t* vt    = (bf16_t*)(ws + 56 * MB);    //  8 MiB: v^T bf16 [16][128][2048]
    float*  qkvf  = (float*) (ws + 64 * MB);    // 64 MiB: qkv f32 (last: absorbs vt tail reads)
    bf16_t* encb  = xb;      // alias: x dead after QKV GEMM
    bf16_t* owt   = wqkvt;   // alias: wqkvt dead after QKV GEMM

    const int M  = B_ * T_;        // 4096
    const int NO = 32 * H_;        // 4096 fused qkv columns

    // 1) cast x, pack W^T (K-major)
    cast_f32_bf16<<<4096, 256, 0, stream>>>(x, xb, M * D_);
    pack_qkv_wt<<<dim3(D_ / 32, NO / 32), dim3(32, 8), 0, stream>>>(q_w, kv_w, wqkvt);

    // 2) QKV GEMM: [4096,2048] x [2048,4096] -> f32
    gemm_bf16_tn<<<dim3(NO / GTN, M / GTM), 256, 0, stream>>>(xb, wqkvt, qkvf, M, NO, D_);

    // 3) fused norm + rope + scale -> bf16 q/k ; transpose v -> bf16 v^T
    norm_rope<<<dim3(M, 24), 32, 0, stream>>>(qkvf, qns, kns, qb, kbuf);
    transpose_v<<<dim3(T_ / 32, H_ / 32, B_ * KH_), dim3(32, 8), 0, stream>>>(qkvf, vt);

    // 4) flash attention -> enc bf16 (aliases xb)
    flash_attn<<<(B_ * NH_ * (T_ / 16)) / 8, 256, 0, stream>>>(qb, kbuf, vt, encb);

    // 5) output GEMM: enc [4096,2048] x o_w^T -> f32 out
    transpose_cast<<<dim3(D_ / 32, D_ / 32), dim3(32, 8), 0, stream>>>(o_w, owt, NH_ * H_, D_);
    gemm_bf16_tn<<<dim3(D_ / GTN, M / GTM), 256, 0, stream>>>(encb, owt, out, M, D_, NH_ * H_);
}
